// STA_446676598895
// MI455X (gfx1250) — compile-verified, matrix operand order checked by reading, not run on hardware
//
#include <hip/hip_runtime.h>

// ---------------------------------------------------------------------------
// PCT-style offset attention for MI455X (gfx1250), wave32 + WMMA bf16.
// B=8, C=256, N=2048, 5 modules. Fused 2-pass softmax (never materializes
// the [N,N] attention map); column-L1 renorm factored out of the AV GEMM.
// ---------------------------------------------------------------------------

static constexpr int Bn   = 8;
static constexpr int Cc   = 256;
static constexpr int Nn   = 2048;
static constexpr int KT   = 32;        // WMMA K for bf16
static constexpr int CT   = Cc / KT;   // 8 K-chunks across channels
static constexpr int NT16 = Nn / 16;   // 128 position tiles
static constexpr int CT16 = Cc / 16;   // 16 channel tiles

typedef __attribute__((ext_vector_type(16))) __bf16 v16bf;
typedef __attribute__((ext_vector_type(8)))  __bf16 v8bf;
typedef __attribute__((ext_vector_type(8)))  float  v8f;

__device__ __forceinline__ v8f wmma_bf16(v16bf a, v16bf b, v8f c) {
  // emits v_wmma_f32_16x16x32_bf16 (8-arg form: neg_a, A, neg_b, B, c_mod, C, reuse_a, reuse_b)
  return __builtin_amdgcn_wmma_f32_16x16x32_bf16(false, a, false, b, (short)0, c, false, false);
}

// A/B fragment K-mapping for 16-bit 16x16x32 WMMA (ISA 7.12.2):
// lane half=0: VGPR0..3 hold K=0..7, VGPR4..7 hold K=16..23
// lane half=1: VGPR0..3 hold K=8..15, VGPR4..7 hold K=24..31
// => element e<8  : K = 8*half + e        (contiguous 8)
//    element e>=8 : K = 16 + 8*half + e-8 (contiguous 8)
__device__ __forceinline__ v16bf frag_row_bf16(const __bf16* __restrict__ row,
                                               int kbase, int hf) {
  const v8bf lo = *(const v8bf*)(row + kbase + 8 * hf);
  const v8bf hi = *(const v8bf*)(row + kbase + 16 + 8 * hf);
  v16bf f;
#pragma unroll
  for (int e = 0; e < 8; ++e) { f[e] = lo[e]; f[e + 8] = hi[e]; }
  return f;
}

__device__ __forceinline__ v16bf frag_row_f32(const float* __restrict__ row,
                                              int kbase, int hf) {
  const float* p0 = row + kbase + 8 * hf;
  v16bf f;
#pragma unroll
  for (int e = 0; e < 8; ++e) {
    f[e]     = (__bf16)p0[e];
    f[e + 8] = (__bf16)p0[16 + e];
  }
  return f;
}

__device__ __forceinline__ float sigmf(float x) { return 1.0f / (1.0f + __expf(-x)); }

// ---------------------------------------------------------------------------
// Projection GEMM: O = act(W(CxC) @ X(CxN) + bias)  (one 16x16 tile per wave)
// MODE 0: store bf16 transposed  O^T[pos][o]   (Q and K^T buffers)
// MODE 1: store bf16 direct      O[o][pos] + bias   (V buffer)
// MODE 2: store f32  relu(O+bias)[o][pos] + Res[o][pos]  (final T stage)
// ---------------------------------------------------------------------------
template <int MODE>
__global__ void __launch_bounds__(256)
proj_kernel(const float* __restrict__ X, long xOff, long xStrideB,
            const float* __restrict__ W, const float* __restrict__ bias,
            __bf16* __restrict__ Obf, float* __restrict__ Of,
            const float* __restrict__ Res, long rOff) {
  __shared__ float lds[8][KT * 16];   // per-wave 32(c) x 16(pos) f32 staging
  const int tid  = threadIdx.x;
  const int wv   = tid >> 5;
  const int lane = tid & 31;
  const int ln   = lane & 15;
  const int hf   = lane >> 4;
  const int t    = blockIdx.x * 8 + wv;      // global wave/tile id
  const int b    = t >> 11;                  // / (CT16*NT16) = /2048
  const int rem  = t & 2047;
  const int otile = rem >> 7;                // 16 output-channel tiles
  const int ptile = rem & 127;               // 128 position tiles

  const float* Xb   = X + (long)b * xStrideB + xOff;
  const float* wrow = W + (long)(otile * 16 + ln) * Cc;

  v8f acc = {};
  for (int kc = 0; kc < CT; ++kc) {
    // coalesced stage: lane loads one c-row (16 floats) of the 32x16 X tile
    const float* xr = Xb + (long)(kc * KT + lane) * Nn + ptile * 16;
    float4 x0 = ((const float4*)xr)[0];
    float4 x1 = ((const float4*)xr)[1];
    float4 x2 = ((const float4*)xr)[2];
    float4 x3 = ((const float4*)xr)[3];
    float* l = &lds[wv][lane * 16];
    ((float4*)l)[0] = x0; ((float4*)l)[1] = x1;
    ((float4*)l)[2] = x2; ((float4*)l)[3] = x3;
    __syncthreads();

    v16bf afr = frag_row_f32(wrow, kc * KT, hf);   // A = W rows (o x c)
    v16bf bfr;                                     // B = X cols (c x pos)
#pragma unroll
    for (int e = 0; e < 8; ++e) {
      bfr[e]     = (__bf16)lds[wv][(8 * hf + e) * 16 + ln];
      bfr[e + 8] = (__bf16)lds[wv][(16 + 8 * hf + e) * 16 + ln];
    }
    acc = wmma_bf16(afr, bfr, acc);
    __syncthreads();
  }

  const int pos = ptile * 16 + ln;
  if constexpr (MODE == 0) {
    __bf16* Ob = Obf + (long)b * Nn * Cc;
#pragma unroll
    for (int r = 0; r < 8; ++r) {
      const int o = otile * 16 + 8 * hf + r;
      Ob[(long)pos * Cc + o] = (__bf16)acc[r];
    }
  } else if constexpr (MODE == 1) {
    __bf16* Ob = Obf + (long)b * Cc * Nn;
#pragma unroll
    for (int r = 0; r < 8; ++r) {
      const int o = otile * 16 + 8 * hf + r;
      Ob[(long)o * Nn + pos] = (__bf16)(acc[r] + bias[o]);
    }
  } else {
    const float* Rb = Res + (long)b * 3 * Cc * Nn + rOff;
    float* Ob = Of + (long)b * Cc * Nn;
#pragma unroll
    for (int r = 0; r < 8; ++r) {
      const int o = otile * 16 + 8 * hf + r;
      float v = fmaxf(acc[r] + bias[o], 0.0f);
      Ob[(long)o * Nn + pos] = v + Rb[(long)o * Nn + pos];
    }
  }
}

// ---------------------------------------------------------------------------
// Pass 1: per-row softmax stats over S = Q K (one 16-row n-tile per wave).
// Q[b][n][c] bf16, Kt[b][m][c] bf16 (K stored transposed => contiguous frags).
// ---------------------------------------------------------------------------
__global__ void __launch_bounds__(256)
rowstats_kernel(const __bf16* __restrict__ Q, const __bf16* __restrict__ Kt,
                float* __restrict__ rowmax, float* __restrict__ rowsum) {
  const int tid  = threadIdx.x;
  const int lane = tid & 31;
  const int ln   = lane & 15;
  const int hf   = lane >> 4;
  const int w    = blockIdx.x * 8 + (tid >> 5);
  const int b    = w >> 7;
  const int ntile = w & 127;

  const __bf16* Qb = Q  + (long)b * Nn * Cc;
  const __bf16* Kb = Kt + (long)b * Nn * Cc;

  v16bf qa[CT];
  const __bf16* qrow = Qb + (long)(ntile * 16 + ln) * Cc;
#pragma unroll
  for (int kc = 0; kc < CT; ++kc) qa[kc] = frag_row_bf16(qrow, kc * KT, hf);

  float rmax[8], rsum[8];
#pragma unroll
  for (int r = 0; r < 8; ++r) { rmax[r] = -1e30f; rsum[r] = 0.0f; }

  for (int mt = 0; mt < NT16; ++mt) {
    const __bf16* krow = Kb + (long)(mt * 16 + ln) * Cc;
    v8f s = {};
#pragma unroll
    for (int kc = 0; kc < CT; ++kc)
      s = wmma_bf16(qa[kc], frag_row_bf16(krow, kc * KT, hf), s);
#pragma unroll
    for (int r = 0; r < 8; ++r) {            // online softmax update
      float nm = fmaxf(rmax[r], s[r]);
      rsum[r] = rsum[r] * __expf(rmax[r] - nm) + __expf(s[r] - nm);
      rmax[r] = nm;
    }
  }
  // reduce (max,sum) across the 16 lanes that share each row
#pragma unroll
  for (int r = 0; r < 8; ++r) {
    float m = rmax[r], su = rsum[r];
#pragma unroll
    for (int off = 8; off >= 1; off >>= 1) {
      float om = __shfl_xor(m, off, 32);
      float os = __shfl_xor(su, off, 32);
      float nm = fmaxf(m, om);
      su = su * __expf(m - nm) + os * __expf(om - nm);
      m = nm;
    }
    if (ln == 0) {
      const int n = ntile * 16 + 8 * hf + r;
      rowmax[(long)b * Nn + n] = m;
      rowsum[(long)b * Nn + n] = su;
    }
  }
}

// ---------------------------------------------------------------------------
// Pass 2: fused  t = x - (x_v @ p) / (eps + colsum),  p = softmax rows of S.
// One 16-col m-tile per block (2 waves, each covering half of C).
// S recomputed in registers; P B-fragment built register-locally from the
// two S D-fragments (no cross-lane traffic); colsum accumulated on the fly.
// ---------------------------------------------------------------------------
__global__ void __launch_bounds__(64)
av_kernel(const __bf16* __restrict__ Q, const __bf16* __restrict__ Kt,
          const __bf16* __restrict__ Vf,
          const float* __restrict__ rowmax, const float* __restrict__ rowsum,
          const float* __restrict__ X, long xOff,
          float* __restrict__ tbuf) {
  const int tid  = threadIdx.x;
  const int wv   = tid >> 5;
  const int lane = tid & 31;
  const int ln   = lane & 15;
  const int hf   = lane >> 4;
  const int b     = blockIdx.x >> 7;
  const int mtile = blockIdx.x & 127;

  const __bf16* Qb = Q  + (long)b * Nn * Cc;
  const __bf16* Kb = Kt + (long)b * Nn * Cc;
  const __bf16* Vb = Vf + (long)b * Cc * Nn;
  const float*  rm = rowmax + (long)b * Nn;
  const float*  rs = rowsum + (long)b * Nn;

  v16bf kb[CT];
  const __bf16* krow = Kb + (long)(mtile * 16 + ln) * Cc;
#pragma unroll
  for (int kc = 0; kc < CT; ++kc) kb[kc] = frag_row_bf16(krow, kc * KT, hf);

  v8f zero = {};
  v8f acc[8];
#pragma unroll
  for (int ct = 0; ct < 8; ++ct) acc[ct] = zero;
  float colacc = 0.0f;

  for (int nb = 0; nb < Nn; nb += 32) {
    const __bf16* q0 = Qb + (long)(nb + ln) * Cc;
    const __bf16* q1 = Qb + (long)(nb + 16 + ln) * Cc;
    if (nb + 32 < Nn)  // warm L2/L0 for next Q rows (global_prefetch_b8)
      __builtin_prefetch(q0 + (long)32 * Cc, 0, 0);
    v8f s0 = zero, s1 = zero;
#pragma unroll
    for (int kc = 0; kc < CT; ++kc) {
      s0 = wmma_bf16(frag_row_bf16(q0, kc * KT, hf), kb[kc], s0);
      s1 = wmma_bf16(frag_row_bf16(q1, kc * KT, hf), kb[kc], s1);
    }
    const int n0 = nb + 8 * hf;
    const int n1 = n0 + 16;
    v16bf pf;             // bf16 B-fragment of P, built register-locally
    float cl = 0.0f;
#pragma unroll
    for (int r = 0; r < 8; ++r) {
      const float p0 = __expf(s0[r] - rm[n0 + r]) / rs[n0 + r];
      const float p1 = __expf(s1[r] - rm[n1 + r]) / rs[n1 + r];
      pf[r]     = (__bf16)p0;
      pf[r + 8] = (__bf16)p1;
      cl += p0 + p1;
    }
    colacc += cl;
#pragma unroll
    for (int ct = 0; ct < 8; ++ct) {
      const __bf16* vrow = Vb + (long)((wv * 8 + ct) * 16 + ln) * Nn;
      acc[ct] = wmma_bf16(frag_row_bf16(vrow, nb, hf), pf, acc[ct]);
    }
  }

  // full column sum lives split across lane and lane^16
  const float cs = colacc + __shfl_xor(colacc, 16, 32);
  const float sc = 1.0f / (1e-7f + cs);
  const float* Xb   = X + (long)b * 3 * Cc * Nn + xOff;
  float*       Tbuf = tbuf + (long)b * Cc * Nn;
  const int m = mtile * 16 + ln;
#pragma unroll
  for (int ct = 0; ct < 8; ++ct) {
#pragma unroll
    for (int r = 0; r < 8; ++r) {
      const int c = (wv * 8 + ct) * 16 + 8 * hf + r;
      Tbuf[(long)c * Nn + m] = Xb[(long)c * Nn + m] - acc[ct][r] * sc;
    }
  }
}

// ---------------------------------------------------------------------------
// Final pointwise combine of the 5 attention outputs.
// ---------------------------------------------------------------------------
__global__ void __launch_bounds__(256)
combine_kernel(const float* __restrict__ att, const float* __restrict__ X,
               float* __restrict__ out) {
  const long idx = (long)blockIdx.x * blockDim.x + threadIdx.x;
  const long BCN = (long)Bn * Cc * Nn;
  if (idx >= BCN) return;
  const long CN = (long)Cc * Nn;
  const long b  = idx / CN;
  const long cn = idx - b * CN;
  const float a0 = att[0 * BCN + idx];
  const float a1 = att[1 * BCN + idx];
  const float a2 = att[2 * BCN + idx];
  const float a3 = att[3 * BCN + idx];
  const float a4 = att[4 * BCN + idx];
  const float mid = X[(b * 3 + 1) * CN + cn];
  const float tu = tanhf(a0 * sigmf(a2));
  const float su = tanhf(a1 * sigmf(a3));
  const float gs = sigmf(a4);
  out[idx] = mid + sigmf(tu * gs + su * gs) * tanhf(a4);
}

// ---------------------------------------------------------------------------
extern "C" void kernel_launch(void* const* d_in, const int* in_sizes, int n_in,
                              void* d_out, int out_size, void* d_ws, size_t ws_size,
                              hipStream_t stream) {
  (void)in_sizes; (void)n_in; (void)out_size; (void)ws_size;
  const float* x   = (const float*)d_in[0];
  const float* Wqk = (const float*)d_in[1];
  const float* Vw  = (const float*)d_in[2];
  const float* Vbv = (const float*)d_in[3];
  const float* Tw  = (const float*)d_in[4];
  const float* Tb  = (const float*)d_in[5];
  float* out = (float*)d_out;

  char* ws = (char*)d_ws;
  size_t off = 0;
  auto take = [&](size_t bytes) -> char* {
    char* p = ws + off;
    off += (bytes + 255) & ~(size_t)255;
    return p;
  };
  const long BNC = (long)Bn * Nn * Cc;
  __bf16* Q      = (__bf16*)take((size_t)BNC * 2);
  __bf16* Kt     = (__bf16*)take((size_t)BNC * 2);
  __bf16* Vf     = (__bf16*)take((size_t)BNC * 2);
  float*  rowmax = (float*)take((size_t)Bn * Nn * 4);
  float*  rowsum = (float*)take((size_t)Bn * Nn * 4);
  float*  tbuf   = (float*)take((size_t)BNC * 4);
  float*  attOut = (float*)take((size_t)5 * BNC * 4);

  // module -> (x slice, y slice) of input x[B][3][C][N]: 0=time,1=mid,2=space
  const int sx[5] = {0, 2, 0, 2, 1};
  const int sy[5] = {0, 2, 2, 0, 1};
  const long CN = (long)Cc * Nn;
  const long CC = (long)Cc * Cc;

  const dim3 projGrid((Bn * CT16 * NT16) / 8), projBlk(256);
  const dim3 rsGrid((Bn * NT16) / 8), rsBlk(256);
  const dim3 avGrid(Bn * NT16), avBlk(64);

  for (int i = 0; i < 5; ++i) {
    const float* Wi = Wqk + (long)i * CC;
    // Q = Wqk @ y   (store [n][c]),  K^T = Wqk @ x (store [m][c])
    proj_kernel<0><<<projGrid, projBlk, 0, stream>>>(x, sy[i] * CN, 3 * CN, Wi,
                                                     nullptr, Q, nullptr, nullptr, 0);
    proj_kernel<0><<<projGrid, projBlk, 0, stream>>>(x, sx[i] * CN, 3 * CN, Wi,
                                                     nullptr, Kt, nullptr, nullptr, 0);
    // V = Vw @ x + Vb (store [c][n])
    proj_kernel<1><<<projGrid, projBlk, 0, stream>>>(x, sx[i] * CN, 3 * CN,
                                                     Vw + (long)i * CC, Vbv + (long)i * Cc,
                                                     Vf, nullptr, nullptr, 0);
    rowstats_kernel<<<rsGrid, rsBlk, 0, stream>>>(Q, Kt, rowmax, rowsum);
    av_kernel<<<avGrid, avBlk, 0, stream>>>(Q, Kt, Vf, rowmax, rowsum,
                                            x, sx[i] * CN, tbuf);
    // attOut[i] = x + relu(Tw @ t + Tb)
    proj_kernel<2><<<projGrid, projBlk, 0, stream>>>(tbuf, 0, CN,
                                                     Tw + (long)i * CC, Tb + (long)i * Cc,
                                                     nullptr, attOut + (long)i * BNC,
                                                     x, sx[i] * CN);
  }
  const long total = (long)Bn * CN;
  combine_kernel<<<dim3((unsigned)((total + 255) / 256)), dim3(256), 0, stream>>>(attOut, x, out);
}
